// DecoderBLock_52037823758971
// MI455X (gfx1250) — compile-verified, matrix operand order checked by reading, not run on hardware
//
#include <hip/hip_runtime.h>
#include <hip/hip_bf16.h>
#include <math.h>

typedef __attribute__((ext_vector_type(16))) _Float16 v16h;
typedef __attribute__((ext_vector_type(8)))  _Float16 v8h;
typedef __attribute__((ext_vector_type(8)))  float    v8f;
typedef _Float16 half_t;

#define WSZ 8
#define SHF 4
#define NHEADS 4

// ---------------- WMMA fragment builders (wave32, 16x16x32 f16) ----------------
// Per-lane data is two contiguous 16B runs -> 2x ds_load_b128 each.
// A 16x32 f16: lanes 0-15 row M=lane, K = {0..7, 16..23}; lanes 16-31 row M=lane-16, K = {8..15, 24..31}
__device__ __forceinline__ v16h frag_a(const half_t* p, int row0, int col0, int ld, int lane) {
  int r = row0 + (lane & 15);
  int kb = col0 + ((lane < 16) ? 0 : 8);
  const half_t* q = p + (size_t)r * ld + kb;
  v8h lo = *(const v8h*)(q);
  v8h hi = *(const v8h*)(q + 16);
  return __builtin_shufflevector(lo, hi, 0, 1, 2, 3, 4, 5, 6, 7, 8, 9, 10, 11, 12, 13, 14, 15);
}
// B 32x16 from column-major/transposed storage: B[k][n] = p[(col0+n)*ld + k0+k]
__device__ __forceinline__ v16h frag_bt(const half_t* p, int k0, int col0, int ld, int lane) {
  int c = col0 + (lane & 15);
  int kb = k0 + ((lane < 16) ? 0 : 8);
  const half_t* q = p + (size_t)c * ld + kb;
  v8h lo = *(const v8h*)(q);
  v8h hi = *(const v8h*)(q + 16);
  return __builtin_shufflevector(lo, hi, 0, 1, 2, 3, 4, 5, 6, 7, 8, 9, 10, 11, 12, 13, 14, 15);
}

__device__ __forceinline__ float gelu_f(float v) {
  return 0.5f * v * (1.f + erff(v * 0.70710678118654752f));
}

// ---------------- Generic WMMA GEMM: C[P,N] = A[P,K] * B[K,N] (+bias,+act,+resid,+accum) ----------
// A token-major f16 (lda=K). gather mode: row p decodes to (b,y,x) on gH x gW grid, reads shifted
// pixel (y+sy, x+sx) (zero outside) -- used for dilated conv taps.
__global__ __launch_bounds__(256) void gemm_wmma_kernel(
    const half_t* __restrict__ A, const half_t* __restrict__ Bm,
    const float* __restrict__ bias, const float* __restrict__ resid,
    float* __restrict__ Cf, half_t* __restrict__ Ch,
    int Pdim, int N, int K, int act, int accum,
    int gather, int gH, int gW, int sy, int sx,
    int chStride, int chOff)
{
  __shared__ __align__(16) half_t As[64 * 32];      // [row][k]
  __shared__ __align__(16) half_t BsT[64 * 32];     // [n][k]  (transposed at store)
  int tid = threadIdx.x, lane = tid & 31, wid = tid >> 5;
  int bm = blockIdx.x * 64, bn = blockIdx.y * 64;
  int wm = (wid >> 1) * 16, wn = (wid & 1) * 32;
  v8f acc0 = {}; v8f acc1 = {};

  int la = tid * 8;            // A tile: 64x32 halves
  int arow = la >> 5, acol = la & 31;
  int gp = bm + arow;
  const half_t* aptr = A;
  bool avalid = true;
  if (gather) {
    int hw = gp % (gH * gW); int bb = gp / (gH * gW);
    int y = hw / gW + sy, x = hw % gW + sx;
    avalid = (y >= 0 && y < gH && x >= 0 && x < gW);
    if (avalid) aptr = A + (size_t)(bb * gH * gW + y * gW + x) * K + acol;
  } else {
    aptr = A + (size_t)gp * K + acol;
  }
  int lb = tid * 8;            // B tile: 32x64 halves
  int brow = lb >> 6, bcol = lb & 63;
  const half_t* bptr = Bm + (size_t)brow * N + bn + bcol;

  for (int k0 = 0; k0 < K; k0 += 32) {
    float4 av = make_float4(0.f, 0.f, 0.f, 0.f);
    if (avalid) av = *(const float4*)(aptr + k0);
    *(float4*)(As + arow * 32 + acol) = av;
    // load B row-chunk coalesced, scatter transposed into LDS
    half_t tb[8];
    *(float4*)tb = *(const float4*)(bptr + (size_t)k0 * N);
#pragma unroll
    for (int j = 0; j < 8; ++j) BsT[(bcol + j) * 32 + brow] = tb[j];
    if (k0 + 32 < K) {  // keep L2 ahead of the pipe
      if (avalid) __builtin_prefetch(aptr + k0 + 32, 0, 3);
      __builtin_prefetch(bptr + (size_t)(k0 + 32) * N, 0, 3);
    }
    __syncthreads();
    v16h fa  = frag_a(As, wm, 0, 32, lane);
    v16h fb0 = frag_bt(BsT, 0, wn, 32, lane);
    v16h fb1 = frag_bt(BsT, 0, wn + 16, 32, lane);
    acc0 = __builtin_amdgcn_wmma_f32_16x16x32_f16(false, fa, false, fb0, (short)0, acc0, false, false);
    acc1 = __builtin_amdgcn_wmma_f32_16x16x32_f16(false, fa, false, fb1, (short)0, acc1, false, false);
    __syncthreads();
  }
  int hi8 = (lane >> 4) << 3;
  int cn = lane & 15;
#pragma unroll
  for (int t = 0; t < 2; ++t) {
    v8f acc = t ? acc1 : acc0;
    int col = bn + wn + t * 16 + cn;
#pragma unroll
    for (int r = 0; r < 8; ++r) {
      int row = bm + wm + r + hi8;
      float v = acc[r];
      if (bias)  v += bias[col];
      if (act == 1) v = fmaxf(v, 0.f);
      if (resid) v += resid[(size_t)row * N + col];
      if (accum && Cf) v += Cf[(size_t)row * N + col];
      if (Cf) Cf[(size_t)row * N + col] = v;
      if (Ch) Ch[(size_t)row * chStride + chOff + col] = (half_t)v;
    }
  }
}

// ---------------- Windowed (optionally shifted) cross attention ----------------
// Q/K/V token-major f16 (P,128). One block per window; NH=4 heads, L=64, hd=32.
__global__ __launch_bounds__(256) void win_attn_kernel(
    const half_t* __restrict__ Qg, const half_t* __restrict__ Kg, const half_t* __restrict__ Vg,
    float* __restrict__ aw, half_t* __restrict__ outg,
    int Himg, int Wimg, int shifted)
{
  __shared__ __align__(16) half_t Qs[64 * 128];          // [l][c]
  __shared__ __align__(16) half_t Ks[64 * 128];          // [l][c] (used transposed via frag_bt)
  __shared__ __align__(16) half_t Vt[128 * 64];          // [c][l] (transposed at store)
  __shared__ __align__(16) half_t Ss[NHEADS * 64 * 64];  // [h][l][m]

  int blk = blockIdx.x;
  int wpr = Wimg / WSZ;
  int wpi = (Himg / WSZ) * wpr;
  int b = blk / wpi;
  int wrem = blk % wpi;
  int wy = wrem / wpr, wx = wrem % wpr;
  int tid = threadIdx.x, lane = tid & 31, wid = tid >> 5;

  auto tokOf = [&](int l) {
    int ly = l >> 3, lx = l & 7;
    int ys = wy * WSZ + ly, xs = wx * WSZ + lx;
    if (shifted) { ys += SHF; if (ys >= Himg) ys -= Himg; xs += SHF; if (xs >= Wimg) xs -= Wimg; }
    return (b * Himg + ys) * Wimg + xs;
  };

  { // gather Q/K/V into LDS: 4 threads per token, 32 halves each; V transposed
    int l = tid >> 2, seg = (tid & 3) * 32;
    int tok = tokOf(l);
    const half_t* q = Qg + (size_t)tok * 128 + seg;
    const half_t* k = Kg + (size_t)tok * 128 + seg;
    const half_t* v = Vg + (size_t)tok * 128 + seg;
#pragma unroll
    for (int i = 0; i < 4; ++i) {
      ((float4*)(Qs + l * 128 + seg))[i] = ((const float4*)q)[i];
      ((float4*)(Ks + l * 128 + seg))[i] = ((const float4*)k)[i];
    }
    half_t tv[32];
#pragma unroll
    for (int i = 0; i < 4; ++i) ((float4*)tv)[i] = ((const float4*)v)[i];
#pragma unroll
    for (int j = 0; j < 32; ++j) Vt[(seg + j) * 64 + l] = tv[j];
  }
  __syncthreads();

  int hi8 = (lane >> 4) << 3;
  int cn = lane & 15;
  const float scale = 0.17677669529663687f; // 1/sqrt(32)

  // S = Q * K^T (per head, 4x4 tiles of 16x16, K=32)
  for (int idx = wid; idx < NHEADS * 16; idx += 8) {
    int h = idx >> 4, tm = (idx >> 2) & 3, tn = idx & 3;
    v16h fa = frag_a(Qs, tm * 16, h * 32, 128, lane);
    v16h fb = frag_bt(Ks, h * 32, tn * 16, 128, lane);
    v8f acc = {};
    acc = __builtin_amdgcn_wmma_f32_16x16x32_f16(false, fa, false, fb, (short)0, acc, false, false);
#pragma unroll
    for (int r = 0; r < 8; ++r) {
      int l = tm * 16 + r + hi8;
      int m = tn * 16 + cn;
      float v = acc[r] * scale;
      if (shifted) {
        int yq = wy * WSZ + (l >> 3), xq = wx * WSZ + (l & 7);
        int ym = wy * WSZ + (m >> 3), xm = wx * WSZ + (m & 7);
        int rq = ((yq < Himg - WSZ) ? 0 : (yq < Himg - SHF ? 1 : 2)) * 3 +
                 ((xq < Wimg - WSZ) ? 0 : (xq < Wimg - SHF ? 1 : 2));
        int rm = ((ym < Himg - WSZ) ? 0 : (ym < Himg - SHF ? 1 : 2)) * 3 +
                 ((xm < Wimg - WSZ) ? 0 : (xm < Wimg - SHF ? 1 : 2));
        if (rq != rm) v += -100.f;
      }
      Ss[h * 4096 + l * 64 + m] = (half_t)v;
    }
  }
  __syncthreads();

  // softmax per row (256 rows total), emit attention weights
  {
    int h = tid >> 6, l = tid & 63;
    half_t* row = Ss + h * 4096 + l * 64;
    float mx = -1e30f;
#pragma unroll 4
    for (int m = 0; m < 64; ++m) mx = fmaxf(mx, (float)row[m]);
    float s = 0.f;
    float buf[64];
#pragma unroll 4
    for (int m = 0; m < 64; ++m) { float e = __expf((float)row[m] - mx); buf[m] = e; s += e; }
    float inv = 1.f / s;
    float* awp = aw + (((size_t)blk * NHEADS + h) * 64 + l) * 64;
#pragma unroll 4
    for (int m = 0; m < 64; ++m) {
      float p = buf[m] * inv;
      row[m] = (half_t)p;
      awp[m] = p;
    }
  }
  __syncthreads();

  // O = S * V (per head: 4 m-tiles x 2 n-tiles of 16, K=64 in 2 steps), scatter with win_rev+roll
  for (int idx = wid; idx < NHEADS * 8; idx += 8) {
    int h = idx >> 3, tm = (idx >> 1) & 3, tn = idx & 1;
    v8f acc = {};
#pragma unroll
    for (int kk = 0; kk < 64; kk += 32) {
      v16h fa = frag_a(Ss + h * 4096, tm * 16, kk, 64, lane);
      v16h fb = frag_bt(Vt, kk, h * 32 + tn * 16, 64, lane);
      acc = __builtin_amdgcn_wmma_f32_16x16x32_f16(false, fa, false, fb, (short)0, acc, false, false);
    }
#pragma unroll
    for (int r = 0; r < 8; ++r) {
      int l = tm * 16 + r + hi8;
      int c = h * 32 + tn * 16 + cn;
      int tok = tokOf(l);
      outg[(size_t)tok * 128 + c] = (half_t)acc[r];
    }
  }
}

// ---------------- GroupNorm stats / apply ----------------
__global__ __launch_bounds__(256) void gn_stats_kernel(
    const float* __restrict__ x, float* __restrict__ mean, float* __restrict__ rstd,
    int HW, int C, int cpg)
{
  __shared__ float s1[256], s2[256];
  int groups = C / cpg;
  int bg = blockIdx.x, b = bg / groups, g = bg % groups;
  int n = HW * cpg;
  float sum = 0.f, ss = 0.f;
  for (int i = threadIdx.x; i < n; i += 256) {
    int hw = i / cpg, cc = g * cpg + i % cpg;
    float v = x[((size_t)b * HW + hw) * C + cc];
    sum += v; ss += v * v;
  }
  s1[threadIdx.x] = sum; s2[threadIdx.x] = ss;
  __syncthreads();
  for (int st = 128; st > 0; st >>= 1) {
    if (threadIdx.x < st) { s1[threadIdx.x] += s1[threadIdx.x + st]; s2[threadIdx.x] += s2[threadIdx.x + st]; }
    __syncthreads();
  }
  if (threadIdx.x == 0) {
    float m = s1[0] / n;
    float var = s2[0] / n - m * m;
    mean[bg] = m;
    rstd[bg] = rsqrtf(var + 1e-5f);
  }
}

__global__ __launch_bounds__(256) void gn_apply_kernel(
    const float* __restrict__ x, const float* __restrict__ mean, const float* __restrict__ rstd,
    const float* __restrict__ gamma, const float* __restrict__ beta,
    half_t* __restrict__ outh, int ohStride, int ohOff,
    float* __restrict__ outf, float* __restrict__ outNCHW,
    int Bn, int HW, int C, int groups, int act)
{
  size_t total = (size_t)Bn * HW * C;
  int cpg = C / groups;
  for (size_t i = (size_t)blockIdx.x * blockDim.x + threadIdx.x; i < total;
       i += (size_t)gridDim.x * blockDim.x) {
    int c = (int)(i % C);
    size_t p = i / C;
    int b = (int)(p / HW), hw = (int)(p % HW);
    int bg = b * groups + c / cpg;
    float v = (x[i] - mean[bg]) * rstd[bg] * gamma[c] + beta[c];
    if (act) v = gelu_f(v);
    if (outh) outh[p * ohStride + ohOff + c] = (half_t)v;
    if (outf) outf[i] = v;
    if (outNCHW) outNCHW[((size_t)b * C + c) * HW + hw] = v;
  }
}

// ---------------- misc elementwise kernels ----------------
__global__ __launch_bounds__(256) void pack_kernel(const float* __restrict__ in,
                                                   half_t* __restrict__ out,
                                                   int Bn, int C, int HW) {
  size_t total = (size_t)Bn * C * HW;
  for (size_t i = (size_t)blockIdx.x * blockDim.x + threadIdx.x; i < total;
       i += (size_t)gridDim.x * blockDim.x) {
    int c = (int)(i % C);
    size_t p = i / C;
    int b = (int)(p / HW), hw = (int)(p % HW);
    out[i] = (half_t)in[((size_t)b * C + c) * HW + hw];
  }
}

__global__ __launch_bounds__(256) void wt_kernel(const float* __restrict__ in,
                                                 half_t* __restrict__ out,
                                                 int K, int N, int sO, int sC, int off) {
  int total = K * N;
  for (int i = blockIdx.x * blockDim.x + threadIdx.x; i < total; i += gridDim.x * blockDim.x) {
    int o = i % N, c = i / N;
    out[i] = (half_t)in[(size_t)o * sO + (size_t)c * sC + off];
  }
}

__global__ __launch_bounds__(256) void bilinear_kernel(const float* __restrict__ in,
                                                       half_t* __restrict__ out) {
  // in: (8, 48*48, 128) f32 ; out: (8, 96*96, 128) f16 ; align_corners scale 47/95
  size_t total = (size_t)8 * 9216 * 128;
  for (size_t i = (size_t)blockIdx.x * blockDim.x + threadIdx.x; i < total;
       i += (size_t)gridDim.x * blockDim.x) {
    int c = (int)(i % 128);
    size_t rest = i / 128;
    int xo = (int)(rest % 96); rest /= 96;
    int yo = (int)(rest % 96);
    int b  = (int)(rest / 96);
    float fy = yo * (47.f / 95.f), fx = xo * (47.f / 95.f);
    int y0 = (int)floorf(fy); int y1 = y0 + 1 > 47 ? 47 : y0 + 1; float wy = fy - y0;
    int x0 = (int)floorf(fx); int x1 = x0 + 1 > 47 ? 47 : x0 + 1; float wx = fx - x0;
    const float* base = in + (size_t)b * 2304 * 128;
    float v00 = base[(size_t)(y0 * 48 + x0) * 128 + c];
    float v01 = base[(size_t)(y0 * 48 + x1) * 128 + c];
    float v10 = base[(size_t)(y1 * 48 + x0) * 128 + c];
    float v11 = base[(size_t)(y1 * 48 + x1) * 128 + c];
    float top = v00 * (1.f - wx) + v01 * wx;
    float bot = v10 * (1.f - wx) + v11 * wx;
    out[i] = (half_t)(top * (1.f - wy) + bot * wy);
  }
}

__global__ void gap_kernel(const float* __restrict__ x, float* __restrict__ gap) {
  int i = blockIdx.x * blockDim.x + threadIdx.x;
  if (i >= 8 * 128) return;
  int b = i / 128, c = i % 128;
  float s = 0.f;
  for (int hw = 0; hw < 9216; ++hw) s += x[((size_t)b * 9216 + hw) * 128 + c];
  gap[i] = s / 9216.f;
}

__global__ void gapfc_kernel(const float* __restrict__ gap, const float* __restrict__ wg,
                             float* __restrict__ out) {
  int i = blockIdx.x * blockDim.x + threadIdx.x;
  if (i >= 8 * 64) return;
  int b = i / 64, m = i % 64;
  float s = 0.f;
  for (int c = 0; c < 128; ++c) s += gap[b * 128 + c] * wg[m * 128 + c];
  out[i] = s;
}

__global__ void gapgn_kernel(const float* __restrict__ y, const float* __restrict__ g,
                             const float* __restrict__ be, float* __restrict__ f1) {
  int i = blockIdx.x * blockDim.x + threadIdx.x;
  if (i >= 64) return;
  int b = i / 8, grp = i % 8;
  const float* p = y + b * 64 + grp * 8;
  float m = 0.f;
  for (int j = 0; j < 8; ++j) m += p[j];
  m *= 0.125f;
  float var = 0.f;
  for (int j = 0; j < 8; ++j) { float d = p[j] - m; var += d * d; }
  var *= 0.125f;
  float rs = rsqrtf(var + 1e-5f);
  for (int j = 0; j < 8; ++j) {
    int c = grp * 8 + j;
    f1[b * 64 + c] = gelu_f((p[j] - m) * rs * g[c] + be[c]);
  }
}

__global__ __launch_bounds__(256) void bcast_kernel(const float* __restrict__ f1,
                                                    half_t* __restrict__ cat, int Pn, int HW) {
  size_t total = (size_t)Pn * 64;
  for (size_t i = (size_t)blockIdx.x * blockDim.x + threadIdx.x; i < total;
       i += (size_t)gridDim.x * blockDim.x) {
    int m = (int)(i % 64);
    size_t p = i / 64;
    int b = (int)(p / HW);
    cat[p * 320 + m] = (half_t)f1[b * 64 + m];
  }
}

__global__ __launch_bounds__(256) void copy_slice_kernel(const half_t* __restrict__ src,
                                                         half_t* __restrict__ dst, size_t Pn,
                                                         int ss, int ds, int doff, int Cc) {
  size_t total = Pn * Cc;
  for (size_t i = (size_t)blockIdx.x * blockDim.x + threadIdx.x; i < total;
       i += (size_t)gridDim.x * blockDim.x) {
    int c = (int)(i % Cc);
    size_t p = i / Cc;
    dst[p * ds + doff + c] = src[p * ss + c];
  }
}

// ---------------- host orchestration ----------------
extern "C" void kernel_launch(void* const* d_in, const int* in_sizes, int n_in,
                              void* d_out, int out_size, void* d_ws, size_t ws_size,
                              hipStream_t stream) {
  (void)in_sizes; (void)n_in; (void)out_size; (void)ws_size;
  const int B = 8, H = 96, W = 96, H2 = 48, W2 = 48;
  const int P = B * H * W;      // 73728
  const int P2 = B * H2 * W2;   // 18432

  const float* skip = (const float*)d_in[0];
  const float* xin  = (const float*)d_in[1];
  const float* xf_w = (const float*)d_in[2];
  const float* xf_b = (const float*)d_in[3];
  const float* xf_g = (const float*)d_in[4];
  const float* xf_be = (const float*)d_in[5];
  const float* sf_w = (const float*)d_in[6];
  const float* sf_b = (const float*)d_in[7];
  const float* sf_g = (const float*)d_in[8];
  const float* sf_be = (const float*)d_in[9];
  // attn0: 10..17, attn1: 18..25
  const float* st1_w = (const float*)d_in[26];
  const float* st1_b = (const float*)d_in[27];
  const float* st2_w = (const float*)d_in[28];
  const float* st2_b = (const float*)d_in[29];
  const float* w_gap = (const float*)d_in[30];
  const float* w_1x1 = (const float*)d_in[31];
  const float* w_d[3] = {(const float*)d_in[32], (const float*)d_in[33], (const float*)d_in[34]};
  const float* agn_g = (const float*)d_in[35];
  const float* agn_b = (const float*)d_in[36];
  const float* w_out = (const float*)d_in[37];
  const float* mlp_w = (const float*)d_in[38];
  const float* mlp_b = (const float*)d_in[39];
  const float* mlp_g = (const float*)d_in[40];
  const float* mlp_be = (const float*)d_in[41];

  float* out_f = (float*)d_out;
  float* aw0 = out_f + (size_t)B * 128 * H * W;          // 9437184
  float* aw1 = aw0 + (size_t)1152 * NHEADS * 64 * 64;    // +18874368

  char* ws = (char*)d_ws;
  size_t off = 0;
  auto alloc = [&](size_t bytes) -> char* {
    char* p = ws + off;
    off = (off + bytes + 255) & ~(size_t)255;
    return p;
  };

  half_t* wT_xf  = (half_t*)alloc((size_t)256 * 128 * 2);
  half_t* wT_sf  = (half_t*)alloc((size_t)128 * 128 * 2);
  half_t* wT_q[2], *wT_k[2], *wT_v[2], *wT_o[2];
  for (int i = 0; i < 2; ++i) {
    wT_q[i] = (half_t*)alloc((size_t)128 * 128 * 2);
    wT_k[i] = (half_t*)alloc((size_t)128 * 128 * 2);
    wT_v[i] = (half_t*)alloc((size_t)128 * 128 * 2);
    wT_o[i] = (half_t*)alloc((size_t)128 * 128 * 2);
  }
  half_t* wT_s1  = (half_t*)alloc((size_t)128 * 128 * 2);
  half_t* wT_s2  = (half_t*)alloc((size_t)128 * 128 * 2);
  half_t* wT_1x1 = (half_t*)alloc((size_t)128 * 64 * 2);
  half_t* wT_d   = (half_t*)alloc((size_t)27 * 128 * 64 * 2);
  half_t* wT_out = (half_t*)alloc((size_t)320 * 128 * 2);
  half_t* wT_mlp = (half_t*)alloc((size_t)256 * 128 * 2);

  half_t* x_pc    = (half_t*)alloc((size_t)P2 * 256 * 2);
  half_t* skip_pc = (half_t*)alloc((size_t)P * 128 * 2);
  float*  pre48   = (float*)alloc((size_t)P2 * 128 * 4);
  float*  post48  = (float*)alloc((size_t)P2 * 128 * 4);
  half_t* xup     = (half_t*)alloc((size_t)P * 128 * 2);
  float*  preA    = (float*)alloc((size_t)P * 128 * 4);   // skip pre-GN, later x_final
  half_t* skipprop= (half_t*)alloc((size_t)P * 128 * 2);
  half_t* qbuf    = (half_t*)alloc((size_t)P * 128 * 2);
  half_t* kbuf    = (half_t*)alloc((size_t)P * 128 * 2);
  half_t* vbuf    = (half_t*)alloc((size_t)P * 128 * 2);
  half_t* attnout = (half_t*)alloc((size_t)P * 128 * 2);
  float*  preB    = (float*)alloc((size_t)P * 128 * 4);   // a_pre / b_pre / aspp pre / mlp pre
  float*  resf    = (float*)alloc((size_t)P * 128 * 4);   // b2 f32
  half_t* cataspp = (half_t*)alloc((size_t)P * 320 * 2);
  half_t* cat2    = (half_t*)alloc((size_t)P * 256 * 2);
  float*  gmean   = (float*)alloc(64 * 4);
  float*  grstd   = (float*)alloc(64 * 4);
  float*  gapbuf  = (float*)alloc((size_t)8 * 128 * 4);
  float*  gapfcb  = (float*)alloc((size_t)8 * 64 * 4);
  float*  f1buf   = (float*)alloc((size_t)8 * 64 * 4);

  auto gemm = [&](const half_t* A, const half_t* Bm, const float* bias, const float* resid,
                  float* Cf, half_t* Ch, int Pd, int Nn, int Kk, int act, int accum,
                  int gather, int sy, int sx, int chS, int chO) {
    dim3 g(Pd / 64, Nn / 64);
    gemm_wmma_kernel<<<g, 256, 0, stream>>>(A, Bm, bias, resid, Cf, Ch, Pd, Nn, Kk,
                                            act, accum, gather, H, W, sy, sx, chS, chO);
  };
  auto wt = [&](const float* in, half_t* out, int K, int N, int sO, int sC, int o) {
    int total = K * N;
    wt_kernel<<<(total + 255) / 256, 256, 0, stream>>>(in, out, K, N, sO, sC, o);
  };

  // ---- weight transposes (O,C)->(C,O) f16 ----
  wt(xf_w, wT_xf, 256, 128, 256, 1, 0);
  wt(sf_w, wT_sf, 128, 128, 128, 1, 0);
  for (int i = 0; i < 2; ++i) {
    int base = 10 + i * 8;
    wt((const float*)d_in[base + 0], wT_q[i], 128, 128, 128, 1, 0);
    wt((const float*)d_in[base + 2], wT_k[i], 128, 128, 128, 1, 0);
    wt((const float*)d_in[base + 4], wT_v[i], 128, 128, 128, 1, 0);
    wt((const float*)d_in[base + 6], wT_o[i], 128, 128, 128, 1, 0);
  }
  wt(st1_w, wT_s1, 128, 128, 128, 1, 0);
  wt(st2_w, wT_s2, 128, 128, 128, 1, 0);
  wt(w_1x1, wT_1x1, 128, 64, 128, 1, 0);
  for (int di = 0; di < 3; ++di)
    for (int tap = 0; tap < 9; ++tap)
      wt(w_d[di], wT_d + (size_t)(di * 9 + tap) * 128 * 64, 128, 64, 1152, 9, tap);
  wt(w_out, wT_out, 320, 128, 320, 1, 0);
  wt(mlp_w, wT_mlp, 256, 128, 256, 1, 0);

  // ---- pack inputs to token-major f16 ----
  pack_kernel<<<4096, 256, 0, stream>>>(xin, x_pc, B, 256, H2 * W2);
  pack_kernel<<<4096, 256, 0, stream>>>(skip, skip_pc, B, 128, H * W);

  // ---- x_feed: conv1x1 -> GN -> GELU -> bilinear upsample ----
  gemm(x_pc, wT_xf, xf_b, nullptr, pre48, nullptr, P2, 128, 256, 0, 0, 0, 0, 0, 0, 0);
  gn_stats_kernel<<<64, 256, 0, stream>>>(pre48, gmean, grstd, H2 * W2, 128, 16);
  gn_apply_kernel<<<4096, 256, 0, stream>>>(pre48, gmean, grstd, xf_g, xf_be,
                                            nullptr, 0, 0, post48, nullptr, B, H2 * W2, 128, 8, 1);
  bilinear_kernel<<<4096, 256, 0, stream>>>(post48, xup);

  // ---- skip_feed: conv1x1 -> GN -> GELU ----
  gemm(skip_pc, wT_sf, sf_b, nullptr, preA, nullptr, P, 128, 128, 0, 0, 0, 0, 0, 0, 0);
  gn_stats_kernel<<<64, 256, 0, stream>>>(preA, gmean, grstd, H * W, 128, 16);
  gn_apply_kernel<<<4096, 256, 0, stream>>>(preA, gmean, grstd, sf_g, sf_be,
                                            skipprop, 128, 0, nullptr, nullptr, B, H * W, 128, 8, 1);

  // ---- attn0 (not shifted): Q from xup, K/V from skipprop ----
  gemm(xup, wT_q[0], (const float*)d_in[11], nullptr, nullptr, qbuf, P, 128, 128, 0, 0, 0, 0, 0, 128, 0);
  gemm(skipprop, wT_k[0], (const float*)d_in[13], nullptr, nullptr, kbuf, P, 128, 128, 0, 0, 0, 0, 0, 128, 0);
  gemm(skipprop, wT_v[0], (const float*)d_in[15], nullptr, nullptr, vbuf, P, 128, 128, 0, 0, 0, 0, 0, 128, 0);
  win_attn_kernel<<<1152, 256, 0, stream>>>(qbuf, kbuf, vbuf, aw0, attnout, H, W, 0);
  gemm(attnout, wT_o[0], (const float*)d_in[17], nullptr, preB, xup, P, 128, 128, 0, 0, 0, 0, 0, 128, 0);
  // stabilizer1: a2 = stab(a) + a   (f16 into qbuf)
  gemm(xup, wT_s1, st1_b, preB, nullptr, qbuf, P, 128, 128, 0, 0, 0, 0, 0, 128, 0);

  // ---- attn1 (shifted): Q from a2, K/V from skipprop ----
  gemm(qbuf, wT_q[1], (const float*)d_in[19], nullptr, nullptr, xup, P, 128, 128, 0, 0, 0, 0, 0, 128, 0);
  gemm(skipprop, wT_k[1], (const float*)d_in[21], nullptr, nullptr, kbuf, P, 128, 128, 0, 0, 0, 0, 0, 128, 0);
  gemm(skipprop, wT_v[1], (const float*)d_in[23], nullptr, nullptr, vbuf, P, 128, 128, 0, 0, 0, 0, 0, 128, 0);
  win_attn_kernel<<<1152, 256, 0, stream>>>(xup, kbuf, vbuf, aw1, attnout, H, W, 1);
  gemm(attnout, wT_o[1], (const float*)d_in[25], nullptr, preB, qbuf, P, 128, 128, 0, 0, 0, 0, 0, 128, 0);
  // stabilizer2: b2 = stab(b) + b   (f32 resf, f16 vbuf)
  gemm(qbuf, wT_s2, st2_b, preB, resf, vbuf, P, 128, 128, 0, 0, 0, 0, 0, 128, 0);

  // ---- ASPP ----
  gap_kernel<<<4, 256, 0, stream>>>(resf, gapbuf);
  gapfc_kernel<<<2, 256, 0, stream>>>(gapbuf, w_gap, gapfcb);
  gapgn_kernel<<<1, 64, 0, stream>>>(gapfcb, agn_g, agn_b, f1buf);
  bcast_kernel<<<4096, 256, 0, stream>>>(f1buf, cataspp, P, H * W);
  // f2 = gelu(gn(conv1x1(b2)))
  gemm(vbuf, wT_1x1, nullptr, nullptr, preB, nullptr, P, 64, 128, 0, 0, 0, 0, 0, 0, 0);
  gn_stats_kernel<<<64, 256, 0, stream>>>(preB, gmean, grstd, H * W, 64, 8);
  gn_apply_kernel<<<4096, 256, 0, stream>>>(preB, gmean, grstd, agn_g, agn_b,
                                            cataspp, 320, 64, nullptr, nullptr, B, H * W, 64, 8, 1);
  // dilated convs as 9 shifted-tap WMMA GEMMs each
  const int dil[3] = {3, 5, 7};
  for (int di = 0; di < 3; ++di) {
    for (int tap = 0; tap < 9; ++tap) {
      int sy = (tap / 3 - 1) * dil[di];
      int sx = (tap % 3 - 1) * dil[di];
      gemm(vbuf, wT_d + (size_t)(di * 9 + tap) * 128 * 64, nullptr, nullptr,
           preB, nullptr, P, 64, 128, 0, tap > 0 ? 1 : 0, 1, sy, sx, 0, 0);
    }
    gn_stats_kernel<<<64, 256, 0, stream>>>(preB, gmean, grstd, H * W, 64, 8);
    gn_apply_kernel<<<4096, 256, 0, stream>>>(preB, gmean, grstd, agn_g, agn_b,
                                              cataspp, 320, 128 + di * 64,
                                              nullptr, nullptr, B, H * W, 64, 8, 1);
  }
  // w_out: relu(cat * W) + b2  -> x_final (f32 preA, f16 into cat2[:,0:128])
  gemm(cataspp, wT_out, nullptr, resf, preA, cat2, P, 128, 320, 1, 0, 0, 0, 0, 256, 0);
  copy_slice_kernel<<<4096, 256, 0, stream>>>(skipprop, cat2, (size_t)P, 128, 256, 128, 128);

  // ---- mlp_out: conv1x1 -> GN -> GELU -> NCHW output ----
  gemm(cat2, wT_mlp, mlp_b, nullptr, preB, nullptr, P, 128, 256, 0, 0, 0, 0, 0, 0, 0);
  gn_stats_kernel<<<64, 256, 0, stream>>>(preB, gmean, grstd, H * W, 128, 16);
  gn_apply_kernel<<<4096, 256, 0, stream>>>(preB, gmean, grstd, mlp_g, mlp_be,
                                            nullptr, 0, 0, nullptr, out_f, B, H * W, 128, 8, 1);
}
